// MinGRULayer_46892452937738
// MI455X (gfx1250) — matrix operand order repaired and checked by reading, not compiled, not run on hardware
//
#include <hip/hip_runtime.h>
#include <hip/hip_bf16.h>
#include <cstdint>

// Problem dims (fixed by reference)
#define BB 4
#define TT 4096
#define HH 1024
#define MM (BB*TT)      // 16384 rows
#define KK HH           // 1024 contraction
#define NN HH           // 1024 output features
#define CL 64           // scan chunk length
#define NC (TT/CL)      // 64 chunks

typedef __attribute__((ext_vector_type(16))) __bf16 v16bf;
typedef __attribute__((ext_vector_type(8)))  float  v8f;

union Frag { v16bf v; uint4 q[2]; };

__device__ __forceinline__ unsigned short f2bf(float f) {
    unsigned int u = __float_as_uint(f);
    unsigned int r = u + 0x7FFFu + ((u >> 16) & 1u);   // round-to-nearest-even
    return (unsigned short)(r >> 16);
}

// ---------------- Kernel 1: fused LayerNorm -> bf16 ----------------
__global__ void __launch_bounds__(256)
ln_bf16_kernel(const float* __restrict__ x, const float* __restrict__ gamma,
               const float* __restrict__ beta, unsigned short* __restrict__ out) {
    __shared__ float s1[256];
    __shared__ float s2[256];
    const int row = blockIdx.x;
    const int tid = threadIdx.x;
    const float* xr = x + (size_t)row * HH;
    float4 v = ((const float4*)xr)[tid];
    float s  = v.x + v.y + v.z + v.w;
    float ss = v.x*v.x + v.y*v.y + v.z*v.z + v.w*v.w;
    s1[tid] = s; s2[tid] = ss;
    __syncthreads();
    for (int off = 128; off > 0; off >>= 1) {
        if (tid < off) { s1[tid] += s1[tid+off]; s2[tid] += s2[tid+off]; }
        __syncthreads();
    }
    const float mu   = s1[0] * (1.0f / HH);
    const float var  = s2[0] * (1.0f / HH) - mu * mu;
    const float rstd = rsqrtf(var + 1e-5f);

    float4 g = ((const float4*)gamma)[tid];
    float4 b = ((const float4*)beta)[tid];
    float y0 = (v.x - mu) * rstd * g.x + b.x;
    float y1 = (v.y - mu) * rstd * g.y + b.y;
    float y2 = (v.z - mu) * rstd * g.z + b.z;
    float y3 = (v.w - mu) * rstd * g.w + b.w;
    uint2 p;
    p.x = (unsigned int)f2bf(y0) | ((unsigned int)f2bf(y1) << 16);
    p.y = (unsigned int)f2bf(y2) | ((unsigned int)f2bf(y3) << 16);
    ((uint2*)(out + (size_t)row * HH))[tid] = p;
}

// ---------------- Kernel 2: fp32 -> bf16 weight convert ----------------
__global__ void cvt_bf16_kernel(const float* __restrict__ in,
                                unsigned short* __restrict__ out, int n4) {
    int i = blockIdx.x * blockDim.x + threadIdx.x;
    if (i < n4) {
        float4 v = ((const float4*)in)[i];
        uint2 p;
        p.x = (unsigned int)f2bf(v.x) | ((unsigned int)f2bf(v.y) << 16);
        p.y = (unsigned int)f2bf(v.z) | ((unsigned int)f2bf(v.w) << 16);
        ((uint2*)out)[i] = p;
    }
}

// ---------------- Kernel 3: WMMA GEMM  out[m,n] = sum_k A[m,k]*W[n,k] + bias[n] ----
// Block: 256 threads = 8 wave32. Block tile 128(M) x 128(N).
// Wave (w&3) -> M group of 32 rows, (w>>2) -> N group of 64 cols: 2x4 v8f accumulators.
__global__ void __launch_bounds__(256)
gemm_wmma_kernel(const unsigned short* __restrict__ A,   // [MM,KK] bf16 bits
                 const unsigned short* __restrict__ W,   // [NN,KK] bf16 bits
                 const float* __restrict__ bias,         // [NN]
                 float* __restrict__ out,                // [MM,NN]
                 int applySigmoid) {
    const int lane    = threadIdx.x & 31;
    const int wv      = threadIdx.x >> 5;
    const int laneSub = lane & 15;
    const int hi      = lane >> 4;        // 0 or 1: selects K-half per ISA layout
    const int m0 = blockIdx.x * 128 + (wv & 3) * 32;
    const int n0 = blockIdx.y * 128 + (wv >> 2) * 64;

    v8f acc[2][4];
#pragma unroll
    for (int mi = 0; mi < 2; ++mi)
#pragma unroll
        for (int ni = 0; ni < 4; ++ni)
            acc[mi][ni] = (v8f){0.f,0.f,0.f,0.f,0.f,0.f,0.f,0.f};

    for (int k0 = 0; k0 < KK; k0 += 32) {
        Frag a[2], b[4];
        // A 16x32 bf16 fragment: lane L<16 holds row M=L, K {0..7,16..23};
        // lane L+16 holds same row, K {8..15,24..31}  (ISA 7.12.2 16-bit A layout)
#pragma unroll
        for (int mi = 0; mi < 2; ++mi) {
            const unsigned short* ap = A + (size_t)(m0 + mi*16 + laneSub) * KK + k0;
            a[mi].q[0] = *(const uint4*)(ap + hi * 8);
            a[mi].q[1] = *(const uint4*)(ap + 16 + hi * 8);
        }
        // B 32x16: lane n<16 holds column N=n (= row n0+ni*16+n of W), K {0..7,16..23};
        // lane n+16 holds K {8..15,24..31} (same K order as A).
#pragma unroll
        for (int ni = 0; ni < 4; ++ni) {
            const unsigned short* wp = W + (size_t)(n0 + ni*16 + laneSub) * KK + k0;
            b[ni].q[0] = *(const uint4*)(wp + hi * 8);
            b[ni].q[1] = *(const uint4*)(wp + 16 + hi * 8);
        }
#pragma unroll
        for (int mi = 0; mi < 2; ++mi)
#pragma unroll
            for (int ni = 0; ni < 4; ++ni)
                acc[mi][ni] = __builtin_amdgcn_wmma_f32_16x16x32_bf16(
                    false, a[mi].v, false, b[ni].v,
                    (short)0, acc[mi][ni], false, false);
    }

    // C/D layout: VGPR r, lanes 0-15 -> M=r, lanes 16-31 -> M=8+r; N = lane&15
#pragma unroll
    for (int mi = 0; mi < 2; ++mi)
#pragma unroll
        for (int ni = 0; ni < 4; ++ni) {
            const int col = n0 + ni * 16 + laneSub;
            const float bval = bias[col];
#pragma unroll
            for (int r = 0; r < 8; ++r) {
                const int row = m0 + mi * 16 + hi * 8 + r;
                float v = acc[mi][ni][r] + bval;
                if (applySigmoid) v = 1.0f / (1.0f + __expf(-v));
                out[(size_t)row * NN + col] = v;
            }
        }
}

// ---------------- Scan phase 1: per-chunk affine composition ----------------
// h_chunk_out = Acc * h_chunk_in + Bcc   with Acc = prod(1-z), Bcc accumulated
__global__ void __launch_bounds__(256)
scan_phase1(const float* __restrict__ z, const float* __restrict__ c,
            float* __restrict__ chA, float* __restrict__ chB) {
    const int hq    = blockIdx.x & 3;
    const int chunk = (blockIdx.x >> 2) & (NC - 1);
    const int b     = blockIdx.x >> 8;
    const int h     = hq * 256 + threadIdx.x;
    size_t base = ((size_t)b * TT + (size_t)chunk * CL) * HH + h;
    float Acc = 1.0f, Bcc = 0.0f;
    for (int t = 0; t < CL; ++t) {
        float zt = z[base + (size_t)t * HH];
        float ct = c[base + (size_t)t * HH];
        float om = 1.0f - zt;
        Acc = om * Acc;
        Bcc = om * Bcc + zt * ct;
    }
    size_t idx = ((size_t)b * NC + chunk) * HH + h;
    chA[idx] = Acc;
    chB[idx] = Bcc;
}

// ---------------- Scan phase 2: sequential prefix over chunks ----------------
__global__ void __launch_bounds__(256)
scan_phase2(const float* __restrict__ chA, const float* __restrict__ chB,
            float* __restrict__ hstart) {
    const int tid = blockIdx.x * blockDim.x + threadIdx.x;  // 0..B*H-1
    const int b = tid >> 10;
    const int h = tid & (HH - 1);
    float hr = 0.0f;
    for (int chunk = 0; chunk < NC; ++chunk) {
        size_t idx = ((size_t)b * NC + chunk) * HH + h;
        hstart[idx] = hr;
        hr = chA[idx] * hr + chB[idx];
    }
}

// ---------------- Scan phase 3: replay chunk with known h_start + residual ----
__global__ void __launch_bounds__(256)
scan_phase3(const float* __restrict__ z, const float* __restrict__ c,
            const float* __restrict__ x, const float* __restrict__ hstart,
            float* __restrict__ out) {
    const int hq    = blockIdx.x & 3;
    const int chunk = (blockIdx.x >> 2) & (NC - 1);
    const int b     = blockIdx.x >> 8;
    const int h     = hq * 256 + threadIdx.x;
    size_t base = ((size_t)b * TT + (size_t)chunk * CL) * HH + h;
    float hr = hstart[((size_t)b * NC + chunk) * HH + h];
    for (int t = 0; t < CL; ++t) {
        size_t i = base + (size_t)t * HH;
        float zt = z[i];
        float ct = c[i];
        hr = (1.0f - zt) * hr + zt * ct;
        out[i] = hr + x[i];
    }
}

extern "C" void kernel_launch(void* const* d_in, const int* in_sizes, int n_in,
                              void* d_out, int out_size, void* d_ws, size_t ws_size,
                              hipStream_t stream) {
    (void)in_sizes; (void)n_in; (void)out_size; (void)ws_size;
    const float* x     = (const float*)d_in[0];
    const float* gamma = (const float*)d_in[1];
    const float* beta  = (const float*)d_in[2];
    const float* Wg    = (const float*)d_in[3];
    const float* bg    = (const float*)d_in[4];
    const float* Wc    = (const float*)d_in[5];
    const float* bc    = (const float*)d_in[6];
    float* out = (float*)d_out;

    // Workspace carve-up (~175 MB)
    char* ws = (char*)d_ws;
    unsigned short* normed = (unsigned short*)ws; ws += (size_t)MM * KK * 2;
    unsigned short* wgb    = (unsigned short*)ws; ws += (size_t)NN * KK * 2;
    unsigned short* wcb    = (unsigned short*)ws; ws += (size_t)NN * KK * 2;
    float* zbuf   = (float*)ws; ws += (size_t)MM * NN * 4;
    float* cbuf   = (float*)ws; ws += (size_t)MM * NN * 4;
    float* chA    = (float*)ws; ws += (size_t)BB * NC * HH * 4;
    float* chB    = (float*)ws; ws += (size_t)BB * NC * HH * 4;
    float* hstart = (float*)ws;

    // 1) LayerNorm + bf16 convert
    ln_bf16_kernel<<<MM, 256, 0, stream>>>(x, gamma, beta, normed);
    // 2) Weight converts
    const int n4 = (NN * KK) / 4;
    cvt_bf16_kernel<<<(n4 + 255) / 256, 256, 0, stream>>>(Wg, wgb, n4);
    cvt_bf16_kernel<<<(n4 + 255) / 256, 256, 0, stream>>>(Wc, wcb, n4);
    // 3) WMMA GEMMs: gate (sigmoid fused) and candidate
    dim3 gg(MM / 128, NN / 128);
    gemm_wmma_kernel<<<gg, 256, 0, stream>>>(normed, wgb, bg, zbuf, 1);
    gemm_wmma_kernel<<<gg, 256, 0, stream>>>(normed, wcb, bc, cbuf, 0);
    // 4) Parallel linear-recurrence scan + residual
    scan_phase1<<<BB * NC * 4, 256, 0, stream>>>(zbuf, cbuf, chA, chB);
    scan_phase2<<<(BB * HH) / 256, 256, 0, stream>>>(chA, chB, hstart);
    scan_phase3<<<BB * NC * 4, 256, 0, stream>>>(zbuf, cbuf, x, hstart, out);
}
